// GINLayer_22196390986098
// MI455X (gfx1250) — matrix operand order repaired and checked by reading, not compiled, not run on hardware
//
#include <hip/hip_runtime.h>
#include <hip/hip_bf16.h>
#include <stdint.h>

#define D 128
#define BN_EPS 1e-5f

typedef __attribute__((ext_vector_type(16))) __bf16 v16bf;
typedef __attribute__((ext_vector_type(8)))  __bf16 v8bf;
typedef __attribute__((ext_vector_type(8)))  float  v8f;

// Hardware f32 atomic add (no return) at device scope -> global_atomic_add_f32.
__device__ __forceinline__ void gl_atomic_add_f32(float* p, float v) {
    asm volatile("global_atomic_add_f32 %0, %1, off scope:SCOPE_DEV"
                 :: "v"((unsigned long long)(uintptr_t)p), "v"(v) : "memory");
}

// ---------------- K0: agg = x ; zero BN accumulators ----------------
__global__ __launch_bounds__(256) void k_init(const float4* __restrict__ x,
                                              float4* __restrict__ agg,
                                              float* __restrict__ stats, int total4) {
    int i = blockIdx.x * blockDim.x + threadIdx.x;
    if (i < total4) agg[i] = x[i];
    if (i < 256) stats[i] = 0.f;
}

// ---------------- K1: scatter-add x[src] into agg[dst] ----------------
__global__ __launch_bounds__(256) void k_scatter(const float* __restrict__ x,
                                                 const long long* __restrict__ ei,
                                                 float* __restrict__ agg, int n_edges) {
    int e    = (blockIdx.x * blockDim.x + threadIdx.x) >> 5;
    int lane = threadIdx.x & 31;
    if (e >= n_edges) return;
    long long s = ei[e];
    long long d = ei[(long long)n_edges + e];
    const float* xs = x   + s * D;
    float*       ad = agg + d * D;
#pragma unroll
    for (int k = 0; k < 4; ++k) {
        int c = lane + k * 32;
        gl_atomic_add_f32(ad + c, xs[c]);
    }
}

// ---------------- K2: fused MLP (WMMA bf16) + BN partial sums ----------------
__device__ __forceinline__ v16bf cat8(v8bf lo, v8bf hi) {
    return __builtin_shufflevector(lo, hi, 0,1,2,3,4,5,6,7,8,9,10,11,12,13,14,15);
}

__global__ __launch_bounds__(256) void k_mlp(const float* __restrict__ agg,
                                             const float* __restrict__ W1,
                                             const float* __restrict__ b1,
                                             const float* __restrict__ W2,
                                             const float* __restrict__ b2,
                                             float* __restrict__ h2out,
                                             float* __restrict__ colsum,
                                             float* __restrict__ colsumsq,
                                             int n_nodes) {
    constexpr int WS = 136;                 // padded LDS row stride (halfs), 16B-aligned rows
    __shared__ __bf16 sW1[D * WS];          // W1^T : sW1[n*WS+k] = W1[k][n]
    __shared__ __bf16 sW2[D * WS];          // W2^T
    __shared__ __bf16 sH [128 * WS];        // relu(h1) staging, 16 rows per wave

    const int tid = threadIdx.x;
    for (int i = tid; i < D * D; i += 256) {
        int k = i >> 7, n = i & 127;
        sW1[n * WS + k] = (__bf16)W1[i];
        sW2[n * WS + k] = (__bf16)W2[i];
    }
    __syncthreads();

    const int lane    = tid & 31;
    const int wave    = tid >> 5;
    const int l15     = lane & 15;
    const int halfsel = lane >> 4;          // 0: K-lo group, 1: K-hi group
    const int row0    = blockIdx.x * 128 + wave * 16;

    // ---- A fragments from agg (f32 -> bf16), ISA 16-bit A 16x32 layout ----
    v16bf afrag[4];
    {
        const int  r  = row0 + l15;
        const bool ok = (r < n_nodes);
        const float* rp = agg + (long long)r * D;
        const int kbase = halfsel * 8;
        const float4 z4 = make_float4(0.f, 0.f, 0.f, 0.f);
#pragma unroll
        for (int kt = 0; kt < 4; ++kt) {
            const float* p = rp + kt * 32 + kbase;
            float4 f0 = ok ? *(const float4*)(p +  0) : z4;
            float4 f1 = ok ? *(const float4*)(p +  4) : z4;
            float4 f2 = ok ? *(const float4*)(p + 16) : z4;
            float4 f3 = ok ? *(const float4*)(p + 20) : z4;
            v16bf a;
            a[0]=(__bf16)f0.x; a[1]=(__bf16)f0.y; a[2]=(__bf16)f0.z; a[3]=(__bf16)f0.w;
            a[4]=(__bf16)f1.x; a[5]=(__bf16)f1.y; a[6]=(__bf16)f1.z; a[7]=(__bf16)f1.w;
            a[8]=(__bf16)f2.x; a[9]=(__bf16)f2.y; a[10]=(__bf16)f2.z; a[11]=(__bf16)f2.w;
            a[12]=(__bf16)f3.x; a[13]=(__bf16)f3.y; a[14]=(__bf16)f3.z; a[15]=(__bf16)f3.w;
            afrag[kt] = a;
        }
    }

    // ---- GEMM1: h1 = relu(agg @ W1 + b1), stage to wave-private LDS ----
    const int wbase = wave * 16;
#pragma unroll
    for (int n0 = 0; n0 < 8; ++n0) {
        const int col = n0 * 16 + l15;
        float bv = b1[col];
        v8f acc;
#pragma unroll
        for (int i = 0; i < 8; ++i) acc[i] = bv;
#pragma unroll
        for (int kt = 0; kt < 4; ++kt) {
            const __bf16* bp = &sW1[col * WS + kt * 32 + halfsel * 8];
            v16bf b = cat8(((const v8bf*)bp)[0], ((const v8bf*)bp)[2]);
            acc = __builtin_amdgcn_wmma_f32_16x16x32_bf16(false, afrag[kt], false, b,
                                                          (short)0, acc, false, false);
        }
#pragma unroll
        for (int j = 0; j < 8; ++j) {
            float v = acc[j];
            v = v > 0.f ? v : 0.f;
            sH[(wbase + j + halfsel * 8) * WS + col] = (__bf16)v;
        }
    }
    // sH region is wave-private: same-wave LDS ordering suffices, no barrier.

    // ---- A fragments for GEMM2 from sH ----
    v16bf a2[4];
    {
        const __bf16* rp = &sH[(wbase + l15) * WS];
        const int kbase = halfsel * 8;
#pragma unroll
        for (int kt = 0; kt < 4; ++kt) {
            v8bf lo = *(const v8bf*)(rp + kt * 32 + kbase);
            v8bf hi = *(const v8bf*)(rp + kt * 32 + kbase + 16);
            a2[kt] = cat8(lo, hi);
        }
    }

    // ---- GEMM2: h2 = h1 @ W2 + b2 ; write h2, accumulate BN sums ----
#pragma unroll
    for (int n0 = 0; n0 < 8; ++n0) {
        const int col = n0 * 16 + l15;
        float bv = b2[col];
        v8f acc;
#pragma unroll
        for (int i = 0; i < 8; ++i) acc[i] = bv;
#pragma unroll
        for (int kt = 0; kt < 4; ++kt) {
            const __bf16* bp = &sW2[col * WS + kt * 32 + halfsel * 8];
            v16bf b = cat8(((const v8bf*)bp)[0], ((const v8bf*)bp)[2]);
            acc = __builtin_amdgcn_wmma_f32_16x16x32_bf16(false, a2[kt], false, b,
                                                          (short)0, acc, false, false);
        }
        float s = 0.f, sq = 0.f;
#pragma unroll
        for (int j = 0; j < 8; ++j) {
            int r = row0 + j + halfsel * 8;
            if (r < n_nodes) {
                float v = acc[j];
                h2out[(long long)r * D + col] = v;
                s += v; sq += v * v;
            }
        }
        gl_atomic_add_f32(&colsum[col],   s);
        gl_atomic_add_f32(&colsumsq[col], sq);
    }
}

// ---------------- K3: fold BN stats into scale/shift ----------------
__global__ void k_bnstats(const float* __restrict__ stats,
                          const float* __restrict__ gamma,
                          const float* __restrict__ beta,
                          float* __restrict__ scsh, int n) {
    int c = threadIdx.x;
    float inv_n = 1.0f / (float)n;
    float mean = stats[c] * inv_n;
    float ex2  = stats[128 + c] * inv_n;
    float var  = ex2 - mean * mean;
    float inv  = rsqrtf(var + BN_EPS);
    float sc   = inv * gamma[c];
    scsh[c]       = sc;
    scsh[128 + c] = beta[c] - mean * sc;
}

// ---------------- K4: out = h2*scale + shift + x (in place on d_out) ----------------
__global__ __launch_bounds__(256) void k_finalize(const float4* __restrict__ x,
                                                  float4* __restrict__ out,
                                                  const float* __restrict__ scsh,
                                                  int total4) {
    int i = blockIdx.x * blockDim.x + threadIdx.x;
    if (i >= total4) return;
    int c4 = (i & 31) * 4;
    float4 h = out[i];
    float4 xv = x[i];
    float4 r;
    r.x = h.x * scsh[c4 + 0] + scsh[128 + c4 + 0] + xv.x;
    r.y = h.y * scsh[c4 + 1] + scsh[128 + c4 + 1] + xv.y;
    r.z = h.z * scsh[c4 + 2] + scsh[128 + c4 + 2] + xv.z;
    r.w = h.w * scsh[c4 + 3] + scsh[128 + c4 + 3] + xv.w;
    out[i] = r;
}

extern "C" void kernel_launch(void* const* d_in, const int* in_sizes, int n_in,
                              void* d_out, int out_size, void* d_ws, size_t ws_size,
                              hipStream_t stream) {
    const float*     x     = (const float*)d_in[0];
    const long long* ei    = (const long long*)d_in[1];
    const float*     W1    = (const float*)d_in[2];
    const float*     b1    = (const float*)d_in[3];
    const float*     W2    = (const float*)d_in[4];
    const float*     b2    = (const float*)d_in[5];
    const float*     gamma = (const float*)d_in[6];
    const float*     beta  = (const float*)d_in[7];
    float*           out   = (float*)d_out;

    const int n_nodes = in_sizes[0] / D;
    const int n_edges = in_sizes[1] / 2;

    float* agg   = (float*)d_ws;
    float* stats = agg + (size_t)n_nodes * D;   // 256 floats: colsum | colsumsq
    float* scsh  = stats + 256;                 // 256 floats: scale  | shift

    const int total4 = (n_nodes * D) / 4;

    k_init<<<(total4 + 255) / 256, 256, 0, stream>>>(
        (const float4*)x, (float4*)agg, stats, total4);

    k_scatter<<<(n_edges + 7) / 8, 256, 0, stream>>>(x, ei, agg, n_edges);

    k_mlp<<<(n_nodes + 127) / 128, 256, 0, stream>>>(
        agg, W1, b1, W2, b2, out, stats, stats + 128, n_nodes);

    k_bnstats<<<1, 128, 0, stream>>>(stats, gamma, beta, scsh, n_nodes);

    k_finalize<<<(total4 + 255) / 256, 256, 0, stream>>>(
        (const float4*)x, (float4*)out, scsh, total4);
}
